// ResidualGCNBlock_22136261443948
// MI455X (gfx1250) — matrix operand order repaired and checked by reading, not compiled, not run on hardware
//
#include <hip/hip_runtime.h>

#define NN 50000
#define NE 800000
#define DIM 128
#define BN_EPS 1e-5f

typedef float v2f  __attribute__((ext_vector_type(2)));
typedef float v8f  __attribute__((ext_vector_type(8)));
typedef __bf16 v16bf __attribute__((ext_vector_type(16)));

#if defined(__has_builtin)
#if __has_builtin(__builtin_amdgcn_wmma_f32_16x16x4_f32)
#define USE_F32_WMMA 1
#endif
#endif

// ---------------- degree / normalization precompute ----------------

__global__ void k_zero(float* __restrict__ p, int n) {
    int i = blockIdx.x * blockDim.x + threadIdx.x;
    if (i < n) p[i] = 0.0f;
}

__global__ void k_count_deg(const int* __restrict__ dst, float* __restrict__ deg) {
    int e = blockIdx.x * blockDim.x + threadIdx.x;
    if (e < NE) atomicAdd(&deg[dst[e]], 1.0f);
}

__global__ void k_finalize_deg(const float* __restrict__ deg,
                               float* __restrict__ dinv,
                               float* __restrict__ selfw) {
    int i = blockIdx.x * blockDim.x + threadIdx.x;
    if (i < NN) {
        float d = deg[i] + 1.0f;          // self-loop
        dinv[i]  = rsqrtf(d);
        selfw[i] = 1.0f / d;              // dinv[i]^2
    }
}

__global__ void k_edge_norm(const int* __restrict__ src, const int* __restrict__ dst,
                            const float* __restrict__ dinv, float* __restrict__ norm_e) {
    int e = blockIdx.x * blockDim.x + threadIdx.x;
    if (e < NE) norm_e[e] = dinv[src[e]] * dinv[dst[e]];
}

// ---------------- WMMA GEMM: Out[N,128] = A[N,128] @ W[128,128] ----------------
// One block = 16 rows x 128 cols. A tile staged in LDS (8KB), 8 waves,
// each wave owns a 16x16 output tile, K swept with v_wmma f32 accumulate.

__global__ __launch_bounds__(256)
void k_gemm_wmma(const float* __restrict__ A, const float* __restrict__ W,
                 float* __restrict__ Out) {
    __shared__ float Alds[16 * DIM];

    const int row0 = blockIdx.x * 16;
    const int tid  = threadIdx.x;

    // cooperative load of 16x128 fp32 tile: 512 float4, 2 per thread
    const float4* Ag = (const float4*)(A + (size_t)row0 * DIM);
    float4* Al = (float4*)Alds;
    Al[tid]       = Ag[tid];
    Al[tid + 256] = Ag[tid + 256];
    __syncthreads();

    const int lane = tid & 31;
    const int wave = tid >> 5;       // 0..7 -> column tile
    const int half = lane >> 4;      // 0 or 1 (upper half-wave)
    const int mr   = lane & 15;      // row within A frag / col within B,C frag
    const int col0 = wave * 16;

    v8f c = {};

#ifdef USE_F32_WMMA
    // V_WMMA_F32_16X16X4_F32: A 16x4 (2 VGPR/lane), B 4x16 (2 VGPR/lane)
    for (int k = 0; k < DIM; k += 4) {
        const int kk = k + 2 * half;
        v2f a, b;
        a.x = Alds[mr * DIM + kk];
        a.y = Alds[mr * DIM + kk + 1];
        b.x = W[kk * DIM + col0 + mr];
        b.y = W[(kk + 1) * DIM + col0 + mr];
        c = __builtin_amdgcn_wmma_f32_16x16x4_f32(
                false, a, false, b, (short)0, c, false, false);
    }
#else
    // Fallback: bf16 inputs, f32 accumulate (codegen-confirmed builtin)
    for (int k = 0; k < DIM; k += 32) {
        v16bf a, b;
        #pragma unroll
        for (int j = 0; j < 8; ++j) {
            const int ka = k + ((j >= 4) ? 16 : 0) + 8 * half + 2 * (j & 3);
            a[2 * j]     = (__bf16)Alds[mr * DIM + ka];
            a[2 * j + 1] = (__bf16)Alds[mr * DIM + ka + 1];
            const int kb = k + ((j >= 4) ? 16 : 0) + 8 * half + 2 * (j & 3);
            b[2 * j]     = (__bf16)W[kb * DIM + col0 + mr];
            b[2 * j + 1] = (__bf16)W[(kb + 1) * DIM + col0 + mr];
        }
        c = __builtin_amdgcn_wmma_f32_16x16x32_bf16(
                false, a, false, b, (short)0, c, false, false);
    }
#endif

    // C/D layout: VGPR r -> M = r (lanes 0-15), M = 8+r (lanes 16-31); N = mr
    #pragma unroll
    for (int r = 0; r < 8; ++r) {
        Out[(size_t)(row0 + half * 8 + r) * DIM + col0 + mr] = c[r];
    }
}

// ---------------- message passing ----------------

// agg = h * selfw + bias   (self-loop term + bias)
__global__ void k_init_agg(const float* __restrict__ h, const float* __restrict__ selfw,
                           const float* __restrict__ bias, float* __restrict__ agg) {
    int i = blockIdx.x * blockDim.x + threadIdx.x;
    if (i < NN * DIM) {
        int col = i & (DIM - 1);
        int row = i >> 7;
        agg[i] = h[i] * selfw[row] + bias[col];
    }
}

// 32 lanes per edge; each lane moves one float4 (coalesced 512B row), 4 atomics
__global__ __launch_bounds__(256)
void k_scatter(const int* __restrict__ src, const int* __restrict__ dst,
               const float* __restrict__ norm_e, const float* __restrict__ h,
               float* __restrict__ agg) {
    int gid  = blockIdx.x * blockDim.x + threadIdx.x;
    int e    = gid >> 5;
    int lane = gid & 31;
    if (e >= NE) return;
    int s = src[e], d = dst[e];
    float w = norm_e[e];
    const float4* hv = (const float4*)(h + (size_t)s * DIM);
    float4 m = hv[lane];
    float* out = agg + (size_t)d * DIM + lane * 4;
    atomicAdd(out + 0, m.x * w);
    atomicAdd(out + 1, m.y * w);
    atomicAdd(out + 2, m.z * w);
    atomicAdd(out + 3, m.w * w);
}

// ---------------- batchnorm ----------------

__global__ __launch_bounds__(256)
void k_col_stats(const float* __restrict__ h, float* __restrict__ sum,
                 float* __restrict__ sumsq) {
    __shared__ float s1[256], s2[256];
    const int col = threadIdx.x & (DIM - 1);
    const int sub = threadIdx.x >> 7;    // two rows in flight per block
    float a = 0.0f, b = 0.0f;
    for (int row = blockIdx.x * 2 + sub; row < NN; row += gridDim.x * 2) {
        float v = h[(size_t)row * DIM + col];
        a += v;
        b += v * v;
    }
    s1[threadIdx.x] = a;
    s2[threadIdx.x] = b;
    __syncthreads();
    if (sub == 0) {
        atomicAdd(&sum[col],   s1[threadIdx.x] + s1[threadIdx.x + 128]);
        atomicAdd(&sumsq[col], s2[threadIdx.x] + s2[threadIdx.x + 128]);
    }
}

__global__ void k_bn_finalize(const float* __restrict__ sum, const float* __restrict__ sumsq,
                              float* __restrict__ mu, float* __restrict__ rinv) {
    int c = threadIdx.x;
    if (c < DIM) {
        const float inv_n = 1.0f / (float)NN;
        float m = sum[c] * inv_n;
        float v = sumsq[c] * inv_n - m * m;    // biased variance
        mu[c]   = m;
        rinv[c] = rsqrtf(v + BN_EPS);
    }
}

__global__ void k_bn_relu(const float* __restrict__ h, const float* __restrict__ mu,
                          const float* __restrict__ rinv, const float* __restrict__ gamma,
                          const float* __restrict__ beta, float* __restrict__ out) {
    int i = blockIdx.x * blockDim.x + threadIdx.x;
    if (i < NN * DIM) {
        int c = i & (DIM - 1);
        float v = (h[i] - mu[c]) * rinv[c] * gamma[c] + beta[c];
        out[i] = fmaxf(v, 0.0f);
    }
}

__global__ void k_bn_res_relu(const float* __restrict__ h, const float* __restrict__ mu,
                              const float* __restrict__ rinv, const float* __restrict__ gamma,
                              const float* __restrict__ beta, const float* __restrict__ x,
                              float* __restrict__ out) {
    int i = blockIdx.x * blockDim.x + threadIdx.x;
    if (i < NN * DIM) {
        int c = i & (DIM - 1);
        float v = (h[i] - mu[c]) * rinv[c] * gamma[c] + beta[c] + x[i];
        out[i] = fmaxf(v, 0.0f);
    }
}

// ---------------- launcher ----------------

extern "C" void kernel_launch(void* const* d_in, const int* in_sizes, int n_in,
                              void* d_out, int out_size, void* d_ws, size_t ws_size,
                              hipStream_t stream) {
    (void)in_sizes; (void)n_in; (void)out_size; (void)ws_size;

    const float* x      = (const float*)d_in[0];
    const int*   src    = (const int*)d_in[1];
    const int*   dst    = src + NE;
    const float* W1     = (const float*)d_in[2];
    const float* b1     = (const float*)d_in[3];
    const float* gamma1 = (const float*)d_in[4];
    const float* beta1  = (const float*)d_in[5];
    const float* W2     = (const float*)d_in[6];
    const float* b2     = (const float*)d_in[7];
    const float* gamma2 = (const float*)d_in[8];
    const float* beta2  = (const float*)d_in[9];
    float* out = (float*)d_out;

    // carve workspace
    float* ws     = (float*)d_ws;
    float* deg    = ws;            ws += NN;
    float* dinv   = ws;            ws += NN;
    float* selfw  = ws;            ws += NN;
    float* norm_e = ws;            ws += NE;
    float* sum    = ws;            ws += DIM;   // sum & sumsq contiguous
    float* sumsq  = ws;            ws += DIM;
    float* mu     = ws;            ws += DIM;
    float* rinv   = ws;            ws += DIM;
    float* bufA   = ws;            ws += (size_t)NN * DIM;
    float* bufB   = ws;            ws += (size_t)NN * DIM;
    float* bufC   = ws;            ws += (size_t)NN * DIM;

    const int ND          = NN * DIM;
    const dim3 blk(256);
    const dim3 grid_nd((ND + 255) / 256);
    const dim3 grid_e((NE + 255) / 256);
    const dim3 grid_n((NN + 255) / 256);
    const dim3 grid_gemm(NN / 16);                 // 3125 blocks, 8 waves each
    const dim3 grid_scat((NE * 32) / 256);         // 32 lanes per edge

    // --- shared normalization precompute ---
    k_zero<<<grid_n, blk, 0, stream>>>(deg, NN);
    k_count_deg<<<grid_e, blk, 0, stream>>>(dst, deg);
    k_finalize_deg<<<grid_n, blk, 0, stream>>>(deg, dinv, selfw);
    k_edge_norm<<<grid_e, blk, 0, stream>>>(src, dst, dinv, norm_e);

    // --- layer 1: conv -> BN -> ReLU ---
    k_gemm_wmma<<<grid_gemm, blk, 0, stream>>>(x, W1, bufA);
    k_init_agg<<<grid_nd, blk, 0, stream>>>(bufA, selfw, b1, bufB);
    k_scatter<<<grid_scat, blk, 0, stream>>>(src, dst, norm_e, bufA, bufB);
    k_zero<<<1, 256, 0, stream>>>(sum, 2 * DIM);
    k_col_stats<<<256, blk, 0, stream>>>(bufB, sum, sumsq);
    k_bn_finalize<<<1, DIM, 0, stream>>>(sum, sumsq, mu, rinv);
    k_bn_relu<<<grid_nd, blk, 0, stream>>>(bufB, mu, rinv, gamma1, beta1, bufA);

    // --- layer 2: conv -> BN -> +identity -> ReLU ---
    k_gemm_wmma<<<grid_gemm, blk, 0, stream>>>(bufA, W2, bufB);
    k_init_agg<<<grid_nd, blk, 0, stream>>>(bufB, selfw, b2, bufC);
    k_scatter<<<grid_scat, blk, 0, stream>>>(src, dst, norm_e, bufB, bufC);
    k_zero<<<1, 256, 0, stream>>>(sum, 2 * DIM);
    k_col_stats<<<256, blk, 0, stream>>>(bufC, sum, sumsq);
    k_bn_finalize<<<1, DIM, 0, stream>>>(sum, sumsq, mu, rinv);
    k_bn_res_relu<<<grid_nd, blk, 0, stream>>>(bufC, mu, rinv, gamma2, beta2, x, out);
}
